// LIAF_DVS_61314953117923
// MI455X (gfx1250) — matrix operand order repaired and checked by reading, not compiled, not run on hardware
//
#include <hip/hip_runtime.h>
#include <hip/hip_bf16.h>

#define DECAY  0.3f
#define THRESH 0.5f
#define BATCH  36
#define TSTEPS 60

typedef __attribute__((ext_vector_type(16))) __bf16 v16bf;
typedef __attribute__((ext_vector_type(8)))  __bf16 v8bf;
typedef __attribute__((ext_vector_type(8)))  float  v8f;

// ---------------------------------------------------------------------------
// LIAF cell update: mem = mem*decay*(1-spike) + u ; spike' = mem>th ; out=relu
// ---------------------------------------------------------------------------
__device__ __forceinline__ float liaf_update(float u, float* mem, float* spk, size_t idx) {
  float m = mem[idx], s = spk[idx];
  m = m * DECAY * (1.0f - s) + u;
  spk[idx] = (m > THRESH) ? 1.0f : 0.0f;
  mem[idx] = m;
  return fmaxf(m, 0.0f);
}

__device__ __forceinline__ v8bf zero8() {
  v8bf v;
  #pragma unroll
  for (int i = 0; i < 8; ++i) v[i] = (__bf16)0.0f;
  return v;
}

// ---------------------------------------------------------------------------
// Weight prepack: conv W[O,Cin,3,3] f32 -> bf16 fragment-linear
// dst layout: [ntile][chunk = tap*(Cin/32)+cc][lane][16]
//   B-frag convention: lanes 0-15 carry K=0..15 (elem j -> K=lg*16+j),
//   lanes 16-31 carry K=16..31, N = lane&15.
// ---------------------------------------------------------------------------
__global__ void __launch_bounds__(256) prep_conv_w(const float* __restrict__ W,
                                                   __bf16* __restrict__ dst,
                                                   int Cin, int n) {
  int tid = blockIdx.x * 256 + threadIdx.x;
  if (tid >= n) return;
  int j     = tid & 15;
  int lane  = (tid >> 4) & 31;
  int rest  = tid >> 9;
  int nch   = 9 * Cin / 32;
  int chunk = rest % nch;
  int ntile = rest / nch;
  int lg = lane >> 4, lpos = lane & 15;
  int klocal = lg * 16 + j;
  int cc  = chunk % (Cin >> 5);
  int tap = chunk / (Cin >> 5);
  int c = cc * 32 + klocal;
  int o = ntile * 16 + lpos;
  dst[tid] = (__bf16)W[((size_t)(o * Cin + c)) * 9 + tap];
}

// Wf1[256,8192] f32, reference K order = c*64 + s (NCHW flatten).
// Our activation flatten is NHWC: k' = s*128 + c. Remap while prepacking.
__global__ void __launch_bounds__(256) prep_fc1_w(const float* __restrict__ Wf1,
                                                  __bf16* __restrict__ dst, int n) {
  int tid = blockIdx.x * 256 + threadIdx.x;
  if (tid >= n) return;
  int j    = tid & 15;
  int lane = (tid >> 4) & 31;
  int rest = tid >> 9;
  int chunk = rest & 255;
  int ntile = rest >> 8;
  int lg = lane >> 4, lpos = lane & 15;
  int kp   = chunk * 32 + lg * 16 + j;  // NHWC-flat K
  int nidx = ntile * 16 + lpos;
  int s = kp >> 7;    // spatial 0..63
  int c = kp & 127;   // channel
  dst[tid] = (__bf16)Wf1[(size_t)nidx * 8192 + c * 64 + s];
}

// ---------------------------------------------------------------------------
// Layer 0: binarize + conv3x3 (2->64) + LIAF. K=18 too thin for WMMA -> VALU.
// out NHWC bf16 [B,32,32,64]
// ---------------------------------------------------------------------------
__global__ void __launch_bounds__(256) conv0_liaf(const float* __restrict__ data,
                                                  const float* __restrict__ W0,
                                                  const float* __restrict__ b0,
                                                  float* mem, float* spk,
                                                  __bf16* __restrict__ out, int t) {
  int tid = blockIdx.x * 256 + threadIdx.x;
  if (tid >= BATCH * 32 * 32 * 64) return;
  int o = tid & 63;
  int p = tid >> 6;
  int x = p & 31, y = (p >> 5) & 31, b = p >> 10;
  float acc = b0[o];
  #pragma unroll
  for (int c = 0; c < 2; ++c)
    for (int ky = 0; ky < 3; ++ky) {
      int yy = y + ky - 1;
      if (yy < 0 || yy > 31) continue;
      for (int kx = 0; kx < 3; ++kx) {
        int xx = x + kx - 1;
        if (xx < 0 || xx > 31) continue;
        float v  = data[((((size_t)b * 2 + c) * 32 + yy) * 32 + xx) * TSTEPS + t];
        float xb = (v > 1.0f) ? 1.0f : 0.0f;
        acc += xb * W0[((o * 2 + c) * 3 + ky) * 3 + kx];
      }
    }
  out[tid] = (__bf16)liaf_update(acc, mem, spk, tid);
}

// ---------------------------------------------------------------------------
// Implicit-GEMM conv3x3 (CIN -> 128), bf16 WMMA, LDS-staged A, fused LIAF.
// Workgroup = 16-pixel row segment x 128 out channels (8 waves x 16x16 tile).
// Stage: 3 rows x 18 cols x CIN halo tile -> LDS (zero-padded borders), then a
// branch-free tap*chunk loop: A frags via ds_load_b128, B double-buffered from
// global (prepacked fragment-linear, L2-resident).
// ---------------------------------------------------------------------------
template <int CIN>
__global__ void __launch_bounds__(256) conv_wmma_liaf(const __bf16* __restrict__ in,     // [B,H,W,CIN]
                                                      const __bf16* __restrict__ Wfrag,  // [8][9*CIN/32][32][16]
                                                      const float* __restrict__ bias,    // [128]
                                                      float* mem, float* spk,
                                                      __bf16* __restrict__ out,          // [B,H,W,128]
                                                      int H, int W) {
  constexpr int NCH = 9 * CIN / 32;
  constexpr int TW  = 18;        // 16 pixels + halo
  constexpr int CP  = CIN + 8;   // padded channel pitch (bf16) to spread LDS banks
  __shared__ __bf16 smem[3 * TW * CP];

  int tidb = threadIdx.x;
  int wave = tidb >> 5;
  int lane = tidb & 31;
  int segs = W >> 4;
  int xseg = blockIdx.x % segs;
  int y    = (blockIdx.x / segs) % H;
  int b    = blockIdx.x / (segs * H);
  int mrow  = lane & 15;
  int khalf = lane >> 4;

  // ---- cooperative staging of the halo tile (zero-padded) ----
  constexpr int NC8 = 3 * TW * (CIN / 8);
  for (int ch = tidb; ch < NC8; ch += 256) {
    int c8  = ch % (CIN / 8);
    int col = (ch / (CIN / 8)) % TW;
    int row = ch / ((CIN / 8) * TW);
    int yy = y + row - 1;
    int xx = (xseg << 4) + col - 1;
    v8bf v;
    if (yy >= 0 && yy < H && xx >= 0 && xx < W) {
      v = *(const v8bf*)(in + (((size_t)b * H + yy) * W + xx) * CIN + c8 * 8);
    } else {
      v = zero8();
    }
    *(v8bf*)(smem + ((row * TW + col) * CP) + c8 * 8) = v;
  }
  __syncthreads();

  // ---- K loop: A from LDS, B double-buffered from global ----
  const __bf16* wbase = Wfrag + ((size_t)wave * NCH * 512) + lane * 16;
  v8bf bw0 = *(const v8bf*)(wbase);
  v8bf bw1 = *(const v8bf*)(wbase + 8);
  v8f acc = {};
  for (int chunk = 0; chunk < NCH; ++chunk) {
    int nxt = (chunk + 1 < NCH) ? (chunk + 1) : chunk;
    v8bf nw0 = *(const v8bf*)(wbase + (size_t)nxt * 512);
    v8bf nw1 = *(const v8bf*)(wbase + (size_t)nxt * 512 + 8);

    int tap = chunk / (CIN / 32);
    int kc  = (chunk % (CIN / 32)) * 32;
    int row = tap / 3;
    int col = mrow + tap % 3;
    const __bf16* abase = smem + ((row * TW + col) * CP) + kc + khalf * 16;

    v16bf afrag;
    *(v8bf*)&afrag         = *(const v8bf*)abase;
    *((v8bf*)&afrag + 1)   = *(const v8bf*)(abase + 8);
    v16bf bfrag;
    *(v8bf*)&bfrag         = bw0;
    *((v8bf*)&bfrag + 1)   = bw1;

    acc = __builtin_amdgcn_wmma_f32_16x16x32_bf16(false, afrag, false, bfrag,
                                                  (short)0, acc, false, false);
    bw0 = nw0;
    bw1 = nw1;
  }

  // C/D layout: VGPR r, lanes<16 -> M=r, lanes>=16 -> M=r+8, N = lane&15
  int ch = (wave << 4) + (lane & 15);
  float bch = bias[ch];
  #pragma unroll
  for (int r = 0; r < 8; ++r) {
    int mm = r + (khalf << 3);
    int px = (xseg << 4) + mm;
    size_t idx = ((((size_t)b * H + y) * W + px) * 128) + ch;
    out[idx] = (__bf16)liaf_update(acc[r] + bch, mem, spk, idx);
  }
}

// ---------------------------------------------------------------------------
// 2x2 avg-pool + LIAF (elementwise). in [B,2H,2W,C] bf16 -> out [B,H,W,C]
// ---------------------------------------------------------------------------
__global__ void __launch_bounds__(256) pool_liaf(const __bf16* __restrict__ in,
                                                 float* mem, float* spk,
                                                 __bf16* __restrict__ out,
                                                 int H, int W, int C, int n) {
  int tid = blockIdx.x * 256 + threadIdx.x;
  if (tid >= n) return;
  int c = tid % C;
  int p = tid / C;
  int x = p % W, y = (p / W) % H, b = p / (W * H);
  int W2 = W * 2;
  size_t base = ((((size_t)b * (H * 2) + 2 * y) * W2 + 2 * x) * C) + c;
  size_t rs = (size_t)W2 * C;
  float u = 0.25f * ((float)in[base] + (float)in[base + C] +
                     (float)in[base + rs] + (float)in[base + rs + C]);
  out[tid] = (__bf16)liaf_update(u, mem, spk, tid);
}

// ---------------------------------------------------------------------------
// FC1: [48(pad),8192] x [8192,256] via WMMA. Input rows 36..47 are zero pad
// living in the memset'd state block -> branch-free K loop, double-buffered.
// 48 wave-tiles = 6 blocks x 8 waves. Fused LIAF, f32 out for fc2.
// ---------------------------------------------------------------------------
__global__ void __launch_bounds__(256) fc1_wmma_liaf(const __bf16* __restrict__ in,     // [48,8192]
                                                     const __bf16* __restrict__ Wfrag,  // [16][256][32][16]
                                                     const float* __restrict__ bias,    // [256]
                                                     float* mem, float* spk,
                                                     float* __restrict__ out) {         // [36,256]
  int wave = threadIdx.x >> 5;
  int lane = threadIdx.x & 31;
  int tile  = blockIdx.x * 8 + wave;  // 0..47
  int ntile = tile & 15;
  int mtile = tile >> 4;
  int mrow  = mtile * 16 + (lane & 15);
  int khalf = lane >> 4;

  const __bf16* abase = in + (size_t)mrow * 8192 + khalf * 16;
  const __bf16* wbase = Wfrag + ((size_t)ntile * 256 * 512) + lane * 16;

  v8bf alo = *(const v8bf*)(abase);
  v8bf ahi = *(const v8bf*)(abase + 8);
  v8bf bw0 = *(const v8bf*)(wbase);
  v8bf bw1 = *(const v8bf*)(wbase + 8);

  v8f acc = {};
  for (int chunk = 0; chunk < 256; ++chunk) {
    int nxt = (chunk + 1 < 256) ? (chunk + 1) : chunk;
    v8bf nalo = *(const v8bf*)(abase + nxt * 32);
    v8bf nahi = *(const v8bf*)(abase + nxt * 32 + 8);
    v8bf nw0  = *(const v8bf*)(wbase + (size_t)nxt * 512);
    v8bf nw1  = *(const v8bf*)(wbase + (size_t)nxt * 512 + 8);

    v16bf afrag;
    *(v8bf*)&afrag       = alo;
    *((v8bf*)&afrag + 1) = ahi;
    v16bf bfrag;
    *(v8bf*)&bfrag       = bw0;
    *((v8bf*)&bfrag + 1) = bw1;

    acc = __builtin_amdgcn_wmma_f32_16x16x32_bf16(false, afrag, false, bfrag,
                                                  (short)0, acc, false, false);
    alo = nalo; ahi = nahi; bw0 = nw0; bw1 = nw1;
  }

  int ch = ntile * 16 + (lane & 15);
  float bch = bias[ch];
  #pragma unroll
  for (int r = 0; r < 8; ++r) {
    int row = mtile * 16 + r + (khalf << 3);
    if (row < BATCH) {
      size_t idx = (size_t)row * 256 + ch;
      out[idx] = liaf_update(acc[r] + bch, mem, spk, idx);
    }
  }
}

// ---------------------------------------------------------------------------
// FC2 (256->11) + LIAF + output-spike accumulation. Tiny -> scalar.
// ---------------------------------------------------------------------------
__global__ void __launch_bounds__(256) fc2_liaf_acc(const float* __restrict__ h1,
                                                    const float* __restrict__ Wf2,
                                                    const float* __restrict__ bf2,
                                                    float* mem, float* spk,
                                                    float* __restrict__ accum) {
  int tid = blockIdx.x * 256 + threadIdx.x;
  if (tid >= BATCH * 11) return;
  int o = tid % 11, b = tid / 11;
  float u = bf2[o];
  const float* hr = h1 + (size_t)b * 256;
  const float* wr = Wf2 + (size_t)o * 256;
  #pragma unroll 4
  for (int k = 0; k < 256; ++k) u += hr[k] * wr[k];
  accum[tid] += liaf_update(u, mem, spk, tid);
}

__global__ void __launch_bounds__(256) scale_out(const float* __restrict__ accum,
                                                 float* __restrict__ out, int n) {
  int tid = blockIdx.x * 256 + threadIdx.x;
  if (tid < n) out[tid] = accum[tid] * (1.0f / (float)TSTEPS);
}

// ---------------------------------------------------------------------------
// Host driver
// ---------------------------------------------------------------------------
extern "C" void kernel_launch(void* const* d_in, const int* in_sizes, int n_in,
                              void* d_out, int out_size, void* d_ws, size_t ws_size,
                              hipStream_t stream) {
  (void)in_sizes; (void)n_in; (void)out_size; (void)ws_size;
  const float* data = (const float*)d_in[0];
  // d_in[1] = win (scalar, fixed 60)
  const float* W0  = (const float*)d_in[2];
  const float* b0  = (const float*)d_in[3];
  const float* W1  = (const float*)d_in[4];
  const float* b1  = (const float*)d_in[5];
  const float* W2  = (const float*)d_in[6];
  const float* b2  = (const float*)d_in[7];
  const float* Wf1 = (const float*)d_in[8];
  const float* bf1 = (const float*)d_in[9];
  const float* Wf2 = (const float*)d_in[10];
  const float* bf2 = (const float*)d_in[11];
  float* outp = (float*)d_out;

  char* ws = (char*)d_ws;
  size_t off = 0;
  auto take = [&](size_t bytes) {
    size_t o = off;
    off = (off + bytes + 255) & ~(size_t)255;
    return o;
  };

  // Prepacked bf16 weights (fragment-linear)
  __bf16* W1p  = (__bf16*)(ws + take((size_t)8 * 18 * 32 * 16 * 2));
  __bf16* W2p  = (__bf16*)(ws + take((size_t)8 * 36 * 32 * 16 * 2));
  __bf16* Wf1p = (__bf16*)(ws + take((size_t)16 * 256 * 32 * 16 * 2));

  // Inter-layer activations (NHWC bf16)
  __bf16* act0  = (__bf16*)(ws + take((size_t)BATCH * 1024 * 64 * 2));
  __bf16* act1  = (__bf16*)(ws + take((size_t)BATCH * 1024 * 128 * 2));
  __bf16* actp1 = (__bf16*)(ws + take((size_t)BATCH * 256 * 128 * 2));
  __bf16* act2  = (__bf16*)(ws + take((size_t)BATCH * 256 * 128 * 2));
  float*  h1out = (float*)(ws + take((size_t)BATCH * 256 * 4));

  // Zeroed-every-call block: LIAF states + fc1 input (rows 36..47 stay zero)
  size_t st0 = off;
  __bf16* actp2 = (__bf16*)(ws + take((size_t)48 * 8192 * 2));  // [48,8192] padded
  float* c0m = (float*)(ws + take((size_t)BATCH * 1024 * 64 * 4));
  float* c0s = (float*)(ws + take((size_t)BATCH * 1024 * 64 * 4));
  float* c1m = (float*)(ws + take((size_t)BATCH * 1024 * 128 * 4));
  float* c1s = (float*)(ws + take((size_t)BATCH * 1024 * 128 * 4));
  float* p1m = (float*)(ws + take((size_t)BATCH * 256 * 128 * 4));
  float* p1s = (float*)(ws + take((size_t)BATCH * 256 * 128 * 4));
  float* c2m = (float*)(ws + take((size_t)BATCH * 256 * 128 * 4));
  float* c2s = (float*)(ws + take((size_t)BATCH * 256 * 128 * 4));
  float* p2m = (float*)(ws + take((size_t)BATCH * 64 * 128 * 4));
  float* p2s = (float*)(ws + take((size_t)BATCH * 64 * 128 * 4));
  float* h1m = (float*)(ws + take((size_t)BATCH * 256 * 4));
  float* h1s = (float*)(ws + take((size_t)BATCH * 256 * 4));
  float* h2m = (float*)(ws + take((size_t)BATCH * 11 * 4));
  float* h2s = (float*)(ws + take((size_t)BATCH * 11 * 4));
  float* accum = (float*)(ws + take((size_t)BATCH * 11 * 4));
  size_t stBytes = off - st0;
  hipMemsetAsync(ws + st0, 0, stBytes, stream);

  // Weight prepack (deterministic, every call)
  {
    int n1 = 8 * 18 * 512;
    prep_conv_w<<<(n1 + 255) / 256, 256, 0, stream>>>(W1, W1p, 64, n1);
    int n2 = 8 * 36 * 512;
    prep_conv_w<<<(n2 + 255) / 256, 256, 0, stream>>>(W2, W2p, 128, n2);
    int n3 = 16 * 256 * 512;
    prep_fc1_w<<<(n3 + 255) / 256, 256, 0, stream>>>(Wf1, Wf1p, n3);
  }

  const int nConv0 = BATCH * 1024 * 64;
  const int nPool1 = BATCH * 16 * 16 * 128;
  const int nPool2 = BATCH * 8 * 8 * 128;

  for (int t = 0; t < TSTEPS; ++t) {
    conv0_liaf<<<(nConv0 + 255) / 256, 256, 0, stream>>>(data, W0, b0, c0m, c0s, act0, t);
    conv_wmma_liaf<64><<<BATCH * 32 * 2, 256, 0, stream>>>(act0, W1p, b1, c1m, c1s, act1, 32, 32);
    pool_liaf<<<(nPool1 + 255) / 256, 256, 0, stream>>>(act1, p1m, p1s, actp1, 16, 16, 128, nPool1);
    conv_wmma_liaf<128><<<BATCH * 16 * 1, 256, 0, stream>>>(actp1, W2p, b2, c2m, c2s, act2, 16, 16);
    pool_liaf<<<(nPool2 + 255) / 256, 256, 0, stream>>>(act2, p2m, p2s, actp2, 8, 8, 128, nPool2);
    fc1_wmma_liaf<<<6, 256, 0, stream>>>(actp2, Wf1p, bf1, h1m, h1s, h1out);
    fc2_liaf_acc<<<2, 256, 0, stream>>>(h1out, Wf2, bf2, h2m, h2s, accum);
  }
  scale_out<<<2, 256, 0, stream>>>(accum, outp, BATCH * 11);
}
